// SelfAttention_44676249813378
// MI455X (gfx1250) — matrix operand order repaired and checked by reading, not compiled, not run on hardware
//
#include <hip/hip_runtime.h>
#include <hip/hip_bf16.h>

// ---------------------------------------------------------------------------
// SelfAttention (B=8, N=E=2048), fp32 in/out.
//   Q = X @ W^T + b; S = Q @ X^T + gamma; attn = softmax(S, axis=1);
//   out[n,m] = sum_k X[n,k]*attn[m,k]
// All matmuls are C = A @ B^T on one WMMA kernel (v_wmma_f32_16x16x32_f16).
// fp32 operands are pre-split once into hi/lo _Float16 arrays (x = hi+lo);
// hi*hi + hi*lo + lo*hi gives ~fp32 accuracy at 8/3 the f32-WMMA rate.
// Global->LDS operand movement uses the gfx1250 Tensor Data Mover
// (tensor_load_to_lds, TENSORcnt) with double-buffered LDS: wave 0 issues 4
// 2D tile descriptors (128x32 f16, HW-padded to 80B LDS rows) for K-step k+1
// while all waves run WMMA on K-step k. ~412 GFLOP, compute-bound.
// ---------------------------------------------------------------------------

typedef __attribute__((ext_vector_type(16))) _Float16 v16h;
typedef __attribute__((ext_vector_type(8)))  _Float16 v8h;
typedef __attribute__((ext_vector_type(4)))  _Float16 v4h;
typedef __attribute__((ext_vector_type(8)))  float    v8f;

// TDM descriptor groups (6-arg clang-23 builtin signature:
// (uint32x4 g0, int32x8 g1, int32x4, int32x4, int32x8, i32 cpol)).
typedef __attribute__((ext_vector_type(4))) unsigned int tdm_g0_t;
typedef __attribute__((ext_vector_type(8))) int          tdm_g1_t;
typedef __attribute__((ext_vector_type(4))) int          tdm_v4i_t;
typedef __attribute__((ext_vector_type(8))) int          tdm_v8i_t;

#define TILE   128
#define TK     32
#define LDSK   40   // 32 data halves + 8 pad halves -> 80B row stride

// Low 32 bits of a generic pointer to __shared__ = LDS byte offset
// (flat LDS aperture).
__device__ inline unsigned lds_off(const void* p) {
    return (unsigned)(unsigned long long)p;
}

// Issue one TDM 2D tile load: 128 rows x 32 f16 from row-major [rows,K] f16
// at gsrc, into LDS at ldsOff with 16B pad after every 64B row (ISA 8.3-8.6).
__device__ inline void tdm_tile_load(const _Float16* gsrc, unsigned ldsOff, int K) {
    const unsigned long long ga = (unsigned long long)gsrc;
    tdm_g0_t g0;
    g0[0] = 1u;                                      // count=1, user desc
    g0[1] = ldsOff;                                  // lds_addr
    g0[2] = (unsigned)ga;                            // global_addr[31:0]
    g0[3] = ((unsigned)(ga >> 32) & 0x1FFFFFFu)      // global_addr[56:32]
          | (2u << 30);                              // type=2 ("image")
    tdm_g1_t g1;
    // wg_mask=0 | data_size=1(2B)<<16 | pad_enable<<20 |
    // pad_interval=3(16 DW)<<22 | pad_amount=3(4 DW=16B)<<25
    g1[0] = (1 << 16) | (1 << 20) | (3 << 22) | (3 << 25);
    g1[1] = (int)(((unsigned)K & 0xFFFFu) << 16);    // tensor_dim0[15:0]
    g1[2] = (int)((((unsigned)K >> 16) & 0xFFFFu)    // tensor_dim0[31:16]
          | (((unsigned)K & 0xFFFFu) << 16));        // tensor_dim1[15:0] (=K rows, OOB only)
    g1[3] = (int)(32u << 16);                        // tensor_dim1[31:16]=0 | tile_dim0=32
    g1[4] = 128;                                     // tile_dim1=128, tile_dim2=0
    g1[5] = K;                                       // tensor_dim0_stride[31:0]
    g1[6] = 0;                                       // stride0[47:32]=0, stride1 lo=0
    g1[7] = 0;
    tdm_v4i_t z4 = {0, 0, 0, 0};                     // groups 2/3 unused (2D tile)
    tdm_v8i_t z8 = {0, 0, 0, 0, 0, 0, 0, 0};         // trailing group (unused)
    __builtin_amdgcn_tensor_load_to_lds(g0, g1, z4, z4, z8, 0);
}

// A-operand fragment (16x32 f16): lane = M%16; elems 0..7 -> K=h*8..,
// elems 8..15 -> K=16+h*8.. (ISA 7.12.2 16-bit A table)
__device__ inline v16h frag_a(const _Float16* rowPtr, int h) {
    v8h r0 = *(const v8h*)(rowPtr + h * 8);
    v8h r1 = *(const v8h*)(rowPtr + 16 + h * 8);
    v16h f;
#pragma unroll
    for (int i = 0; i < 8; ++i) { f[i] = r0[i]; f[i + 8] = r1[i]; }
    return f;
}

// B-operand fragment (32x16 f16): lane = N%16; elems 0..7 -> K=h*8..,
// elems 8..15 -> K=16+h*8.. (same k order as A).
__device__ inline v16h frag_b(const _Float16* rowPtr, int h) {
    v8h r0 = *(const v8h*)(rowPtr + h * 8);
    v8h r1 = *(const v8h*)(rowPtr + 16 + h * 8);
    v16h f;
#pragma unroll
    for (int i = 0; i < 8; ++i) { f[i] = r0[i]; f[i + 8] = r1[i]; }
    return f;
}

__device__ inline void compute_stage(const _Float16 (*sAh)[LDSK], const _Float16 (*sAl)[LDSK],
                                     const _Float16 (*sBh)[LDSK], const _Float16 (*sBl)[LDSK],
                                     int wm, int wn, int lr, int lh, v8f acc[4][2]) {
    v16h ah[4], al[4], bh[2], bl[2];
#pragma unroll
    for (int mi = 0; mi < 4; ++mi) {
        const int r = wm * 64 + mi * 16 + lr;
        ah[mi] = frag_a(&sAh[r][0], lh);
        al[mi] = frag_a(&sAl[r][0], lh);
    }
#pragma unroll
    for (int ni = 0; ni < 2; ++ni) {
        const int r = wn * 32 + ni * 16 + lr;
        bh[ni] = frag_b(&sBh[r][0], lh);
        bl[ni] = frag_b(&sBl[r][0], lh);
    }
    // hi*hi pass, then hi*lo, then lo*hi: spaces accumulator RAW chains.
#pragma unroll
    for (int mi = 0; mi < 4; ++mi)
#pragma unroll
        for (int ni = 0; ni < 2; ++ni)
            acc[mi][ni] = __builtin_amdgcn_wmma_f32_16x16x32_f16(
                false, ah[mi], false, bh[ni], (short)0, acc[mi][ni], false, false);
#pragma unroll
    for (int mi = 0; mi < 4; ++mi)
#pragma unroll
        for (int ni = 0; ni < 2; ++ni)
            acc[mi][ni] = __builtin_amdgcn_wmma_f32_16x16x32_f16(
                false, ah[mi], false, bl[ni], (short)0, acc[mi][ni], false, false);
#pragma unroll
    for (int mi = 0; mi < 4; ++mi)
#pragma unroll
        for (int ni = 0; ni < 2; ++ni)
            acc[mi][ni] = __builtin_amdgcn_wmma_f32_16x16x32_f16(
                false, al[mi], false, bh[ni], (short)0, acc[mi][ni], false, false);
}

// C = A @ B^T with A,B given as hi/lo f16 pairs, row-major [rows,K].
// Output: fp32 C (with optional bias[col] and *scalarAdd), or split f16
// Chi/Clo (with optional bias) when C == nullptr.
// Grid: (Ncols/128, M/128, batch). Block: 256 = 8 waves.
__global__ __launch_bounds__(256) void gemm_abt_f16split_tdm(
    const _Float16* __restrict__ Ahi, const _Float16* __restrict__ Alo, long long strideA,
    const _Float16* __restrict__ Bhi, const _Float16* __restrict__ Blo, long long strideB,
    float* __restrict__ C, _Float16* __restrict__ Chi, _Float16* __restrict__ Clo,
    long long strideC, int K, int ldc,
    const float* __restrict__ bias, const float* __restrict__ scalarAdd)
{
    __shared__ __attribute__((aligned(16))) _Float16 sA[2][2][TILE][LDSK];
    __shared__ __attribute__((aligned(16))) _Float16 sB[2][2][TILE][LDSK];

    const int tid  = threadIdx.x;
    const int lane = tid & 31;
    const int wave = tid >> 5;
    const int wm   = wave >> 2;         // 0..1  (64 rows)
    const int wn   = wave & 3;          // 0..3  (32 cols)
    const int lr   = lane & 15;
    const int lh   = lane >> 4;

    const int rowBase = blockIdx.y * TILE;
    const int colBase = blockIdx.x * TILE;
    const int batch   = blockIdx.z;

    const _Float16* ApH = Ahi + (size_t)batch * strideA + (size_t)rowBase * K;
    const _Float16* ApL = Alo + (size_t)batch * strideA + (size_t)rowBase * K;
    const _Float16* BpH = Bhi + (size_t)batch * strideB + (size_t)colBase * K;
    const _Float16* BpL = Blo + (size_t)batch * strideB + (size_t)colBase * K;

    const bool producer = (wave == 0);
    const int  ksteps   = K / TK;

    v8f acc[4][2];
#pragma unroll
    for (int mi = 0; mi < 4; ++mi)
#pragma unroll
        for (int ni = 0; ni < 2; ++ni)
#pragma unroll
            for (int v = 0; v < 8; ++v) acc[mi][ni][v] = 0.0f;

    if (producer) {                      // prefetch stage 0
        tdm_tile_load(ApH, lds_off(&sA[0][0][0][0]), K);
        tdm_tile_load(ApL, lds_off(&sA[0][1][0][0]), K);
        tdm_tile_load(BpH, lds_off(&sB[0][0][0][0]), K);
        tdm_tile_load(BpL, lds_off(&sB[0][1][0][0]), K);
    }

    for (int kb = 0; kb < ksteps; ++kb) {
        const int st = kb & 1;
        if (producer) {
            if (kb + 1 < ksteps) {       // prefetch next stage, then wait for current
                const int k0 = (kb + 1) * TK;
                const int ns = st ^ 1;
                tdm_tile_load(ApH + k0, lds_off(&sA[ns][0][0][0]), K);
                tdm_tile_load(ApL + k0, lds_off(&sA[ns][1][0][0]), K);
                tdm_tile_load(BpH + k0, lds_off(&sB[ns][0][0][0]), K);
                tdm_tile_load(BpL + k0, lds_off(&sB[ns][1][0][0]), K);
                __builtin_amdgcn_s_wait_tensorcnt(4);
            } else {
                __builtin_amdgcn_s_wait_tensorcnt(0);
            }
        }
        __syncthreads();                 // publish buffer[st]
        compute_stage(sA[st][0], sA[st][1], sB[st][0], sB[st][1], wm, wn, lr, lh, acc);
        __syncthreads();                 // all reads done before buffer reuse
    }

    const float g = (scalarAdd != nullptr) ? *scalarAdd : 0.0f;
    const size_t cBase = (size_t)batch * strideC;
#pragma unroll
    for (int ni = 0; ni < 2; ++ni) {
        const int col  = colBase + wn * 32 + ni * 16 + lr;
        const float ba = (bias != nullptr) ? bias[col] : 0.0f;
#pragma unroll
        for (int mi = 0; mi < 4; ++mi) {
#pragma unroll
            for (int vg = 0; vg < 8; ++vg) {
                // C/D layout: VGPR vg -> M = vg + 8*(lane/16), N = lane%16
                const int row = rowBase + wm * 64 + mi * 16 + vg + 8 * lh;
                const float v = acc[mi][ni][vg] + ba + g;
                const size_t idx = cBase + (size_t)row * ldc + col;
                if (C != nullptr) {
                    C[idx] = v;
                } else {
                    const _Float16 h = (_Float16)v;
                    Chi[idx] = h;
                    Clo[idx] = (_Float16)(v - (float)h);
                }
            }
        }
    }
}

// ------------------- one-time fp32 -> hi/lo f16 split --------------------
__global__ void split_f32_kernel(const float* __restrict__ in,
                                 _Float16* __restrict__ hi, _Float16* __restrict__ lo) {
    const size_t i = ((size_t)blockIdx.x * blockDim.x + threadIdx.x) * 4;
    const float4 v = *(const float4*)(in + i);
    v4h h, l;
    h[0] = (_Float16)v.x; l[0] = (_Float16)(v.x - (float)h[0]);
    h[1] = (_Float16)v.y; l[1] = (_Float16)(v.y - (float)h[1]);
    h[2] = (_Float16)v.z; l[2] = (_Float16)(v.z - (float)h[2]);
    h[3] = (_Float16)v.w; l[3] = (_Float16)(v.w - (float)h[3]);
    *(v4h*)(hi + i) = h;
    *(v4h*)(lo + i) = l;
}

// ---------------- column softmax (axis=1 of S[b,n,m]) --------------------
__global__ void col_softmax_partial(const float* __restrict__ S,
                                    float* __restrict__ Pmax, float* __restrict__ Psum,
                                    int N)
{
    const int b  = blockIdx.z;
    const int rc = blockIdx.y;
    const int c  = blockIdx.x * blockDim.x + threadIdx.x;
    const float* Sb = S + (size_t)b * N * N;
    const int r0 = rc * 128;
    float m = -3.402823466e38f, s = 0.0f;
    for (int r = r0; r < r0 + 128; ++r) {
        const float x  = Sb[(size_t)r * N + c];
        const float nm = fmaxf(m, x);
        s = s * expf(m - nm) + expf(x - nm);
        m = nm;
    }
    const size_t pi = ((size_t)(b * gridDim.y + rc)) * N + c;
    Pmax[pi] = m; Psum[pi] = s;
}

__global__ void col_softmax_finalize(const float* __restrict__ Pmax,
                                     const float* __restrict__ Psum,
                                     float* __restrict__ Mx, float* __restrict__ Sinv,
                                     int N, int chunks)
{
    const int b = blockIdx.y;
    const int c = blockIdx.x * blockDim.x + threadIdx.x;
    float m = -3.402823466e38f, s = 0.0f;
    for (int i = 0; i < chunks; ++i) {
        const size_t pi = ((size_t)(b * chunks + i)) * N + c;
        const float pm = Pmax[pi], ps = Psum[pi];
        const float nm = fmaxf(m, pm);
        s = s * expf(m - nm) + ps * expf(pm - nm);
        m = nm;
    }
    Mx[(size_t)b * N + c]   = m;
    Sinv[(size_t)b * N + c] = 1.0f / s;
}

// attn = exp(S - max[c]) * inv_sum[c], written directly as hi/lo f16 split.
__global__ void col_softmax_apply_split(const float* __restrict__ S,
                                        const float* __restrict__ Mx,
                                        const float* __restrict__ Sinv,
                                        _Float16* __restrict__ Thi,
                                        _Float16* __restrict__ Tlo, int N)
{
    const size_t flat = ((size_t)blockIdx.x * blockDim.x + threadIdx.x) * 4;
    const int c  = (int)(flat % N);
    const size_t bn = flat / N;
    const int b  = (int)(bn / N);
    const float4 x  = *(const float4*)(S + flat);
    const float4 mm = *(const float4*)(Mx   + (size_t)b * N + c);
    const float4 si = *(const float4*)(Sinv + (size_t)b * N + c);
    float e0 = expf(x.x - mm.x) * si.x;
    float e1 = expf(x.y - mm.y) * si.y;
    float e2 = expf(x.z - mm.z) * si.z;
    float e3 = expf(x.w - mm.w) * si.w;
    v4h h, l;
    h[0] = (_Float16)e0; l[0] = (_Float16)(e0 - (float)h[0]);
    h[1] = (_Float16)e1; l[1] = (_Float16)(e1 - (float)h[1]);
    h[2] = (_Float16)e2; l[2] = (_Float16)(e2 - (float)h[2]);
    h[3] = (_Float16)e3; l[3] = (_Float16)(e3 - (float)h[3]);
    *(v4h*)(Thi + flat) = h;
    *(v4h*)(Tlo + flat) = l;
}

// ---------------------------------------------------------------------------
extern "C" void kernel_launch(void* const* d_in, const int* in_sizes, int n_in,
                              void* d_out, int out_size, void* d_ws, size_t ws_size,
                              hipStream_t stream) {
    (void)in_sizes; (void)n_in; (void)out_size; (void)ws_size;
    const int Bb = 8, N = 2048, E = 2048;

    const float* X     = (const float*)d_in[0];  // [B,N,E]
    const float* W     = (const float*)d_in[1];  // [E,E]
    const float* bias  = (const float*)d_in[2];  // [E]
    const float* gamma = (const float*)d_in[3];  // scalar
    float*       out   = (float*)d_out;          // [B,N,N]

    const size_t nBNE = (size_t)Bb * N * E;
    const size_t nEE  = (size_t)E * E;
    const size_t nBNN = (size_t)Bb * N * N;
    const int    chunks = N / 128;

    // Workspace carve (256B-aligned regions).
    char* w = (char*)d_ws;
    auto carve = [&](size_t bytes) -> void* {
        void* p = (void*)w;
        w += (bytes + 255) & ~(size_t)255;
        return p;
    };
    _Float16* Xhi = (_Float16*)carve(nBNE * 2);
    _Float16* Xlo = (_Float16*)carve(nBNE * 2);
    _Float16* Whi = (_Float16*)carve(nEE * 2);
    _Float16* Wlo = (_Float16*)carve(nEE * 2);
    _Float16* Qhi = (_Float16*)carve(nBNE * 2);  // reused for attn split (Thi)
    _Float16* Qlo = (_Float16*)carve(nBNE * 2);  // reused for attn split (Tlo)
    float*    S    = (float*)carve(nBNN * 4);
    float*    Pmax = (float*)carve((size_t)Bb * chunks * N * 4);
    float*    Psum = (float*)carve((size_t)Bb * chunks * N * 4);
    float*    Mx   = (float*)carve((size_t)Bb * N * 4);
    float*    Sinv = (float*)carve((size_t)Bb * N * 4);

    const dim3 blk(256);
    const dim3 gG(E / TILE, N / TILE, Bb);

    // One-time hi/lo splits of the fp32 inputs.
    split_f32_kernel<<<dim3((unsigned)(nBNE / 4 / 256)), blk, 0, stream>>>(X, Xhi, Xlo);
    split_f32_kernel<<<dim3((unsigned)(nEE  / 4 / 256)), blk, 0, stream>>>(W, Whi, Wlo);

    // GEMM1: Q = X @ W^T + bias -> split output (W shared: strideB=0).
    gemm_abt_f16split_tdm<<<gG, blk, 0, stream>>>(
        Xhi, Xlo, (long long)N * E, Whi, Wlo, 0LL,
        nullptr, Qhi, Qlo, (long long)N * E, E, E, bias, nullptr);

    // GEMM2: S = Q @ X^T + gamma -> fp32.
    gemm_abt_f16split_tdm<<<gG, blk, 0, stream>>>(
        Qhi, Qlo, (long long)N * E, Xhi, Xlo, (long long)N * E,
        S, nullptr, nullptr, (long long)N * N, E, N, nullptr, gamma);

    // Column softmax (axis=1); attn written as hi/lo split into Q's buffers.
    col_softmax_partial<<<dim3(N / 256, chunks, Bb), blk, 0, stream>>>(S, Pmax, Psum, N);
    col_softmax_finalize<<<dim3(N / 256, Bb), blk, 0, stream>>>(Pmax, Psum, Mx, Sinv, N, chunks);
    col_softmax_apply_split<<<dim3((unsigned)(nBNN / 4 / 256)), blk, 0, stream>>>(
        S, Mx, Sinv, Qhi, Qlo, N);

    // GEMM3: out = X @ attn^T -> fp32 d_out.
    gemm_abt_f16split_tdm<<<gG, blk, 0, stream>>>(
        Xhi, Xlo, (long long)N * E, Qhi, Qlo, (long long)N * N,
        out, nullptr, nullptr, (long long)N * N, E, N, nullptr, nullptr);
}
